// MultiHeadLatentAttention_73624329388295
// MI455X (gfx1250) — matrix operand order repaired and checked by reading, not compiled, not run on hardware
//
#include <hip/hip_runtime.h>
#include <hip/hip_bf16.h>
#include <stdint.h>

typedef __attribute__((ext_vector_type(16))) __bf16 v16bf;
typedef __attribute__((ext_vector_type(8)))  __bf16 v8bf;
typedef __attribute__((ext_vector_type(8)))  float  v8f;
typedef __attribute__((ext_vector_type(4)))  uint32_t u32x4;
typedef __attribute__((ext_vector_type(8)))  uint32_t u32x8;

#define SEQ 2048
#define ROWS 8192   // B * S

// ---------------------------------------------------------------------------
// f32 -> bf16 conversion (one-time re-precision pass for x and weights)
// ---------------------------------------------------------------------------
__global__ __launch_bounds__(256)
void cvt_f32_bf16(const float* __restrict__ in, __bf16* __restrict__ out, int n) {
  int i = (int)(blockIdx.x * blockDim.x + threadIdx.x);
  if (i < n) out[i] = (__bf16)in[i];
}

// ---------------------------------------------------------------------------
// C = A(MxK,bf16) * W(NxK,bf16)^T ; one wave -> 16 x (16*NSUB) strip.
// A-operand (16x32 bf16): lane L -> row L&15, kg=L>>4; halves j:
//   K = kg*16 + j  => one 32B run per lane (two adjacent 16B loads).
// B-operand (32x16 bf16): lane L -> col L&15; K = (L>>4)*16 + j => one 32B run.
// C/D (16x16 f32): lane L -> col L&15; VGPR i -> row (L>>4)*8 + i.
// ---------------------------------------------------------------------------
template<int NSUB, bool OUT_BF16>
__global__ __launch_bounds__(256)
void gemm_bf16(const __bf16* __restrict__ A, const __bf16* __restrict__ W,
               void* __restrict__ Cp, int M, int N, int K) {
  int gw   = (int)((blockIdx.x * blockDim.x + threadIdx.x) >> 5);
  int lane = (int)(threadIdx.x & 31);
  int groupsN = N / (16 * NSUB);
  int tiles   = (M >> 4) * groupsN;
  if (gw >= tiles) return;
  int tm = gw / groupsN;
  int tg = gw - tm * groupsN;
  int r  = lane & 15;
  int kg = lane >> 4;

  const __bf16* arow  = A + (size_t)(tm * 16 + r) * K;
  const __bf16* wrow0 = W + (size_t)(tg * (16 * NSUB) + r) * K;

  v8f acc[NSUB];
#pragma unroll
  for (int s = 0; s < NSUB; ++s) acc[s] = (v8f){};

  for (int k0 = 0; k0 < K; k0 += 32) {
    __builtin_prefetch((const void*)(arow + k0 + 256), 0, 1);
    __builtin_prefetch((const void*)(wrow0 + k0 + 256), 0, 1);
    v8bf alo = *(const v8bf*)(arow + k0 + kg * 16);
    v8bf ahi = *(const v8bf*)(arow + k0 + kg * 16 + 8);
    v16bf a = __builtin_shufflevector(alo, ahi,
        0,1,2,3,4,5,6,7,8,9,10,11,12,13,14,15);
#pragma unroll
    for (int s = 0; s < NSUB; ++s) {
      v16bf bb = *(const v16bf*)(wrow0 + (size_t)(s * 16) * K + k0 + kg * 16);
      acc[s] = __builtin_amdgcn_wmma_f32_16x16x32_bf16(
          false, a, false, bb, (short)0, acc[s], false, false);
    }
  }
  int mb = tm * 16 + (kg << 3);
  int nb = tg * (16 * NSUB) + r;
  if constexpr (OUT_BF16) {
    __bf16* C = (__bf16*)Cp;
#pragma unroll
    for (int s = 0; s < NSUB; ++s)
#pragma unroll
      for (int i = 0; i < 8; ++i)
        C[(size_t)(mb + i) * N + nb + s * 16] = (__bf16)acc[s][i];
  } else {
    float* C = (float*)Cp;
#pragma unroll
    for (int s = 0; s < NSUB; ++s)
#pragma unroll
      for (int i = 0; i < 8; ++i)
        C[(size_t)(mb + i) * N + nb + s * 16] = acc[s][i];
  }
}

// ---------------------------------------------------------------------------
// In-place RoPE on bf16 rows of (x1,x2) pairs; head dim 32 (16 pairs/head).
// ---------------------------------------------------------------------------
__global__ __launch_bounds__(256)
void rope_bf16(__bf16* __restrict__ data, int rows, int pairsPerRow) {
  int idx = (int)(blockIdx.x * blockDim.x + threadIdx.x);
  int total = rows * pairsPerRow;
  if (idx >= total) return;
  int row = idx / pairsPerRow;
  int p   = idx - row * pairsPerRow;
  int j   = p & 15;
  int pos = row & (SEQ - 1);
  float inv = __expf(-(float)(2 * j) * (1.0f / 32.0f) * 9.210340371976184f);
  float ang = (float)pos * inv;
  float c = cosf(ang), s = sinf(ang);
  __bf16* pd = data + (size_t)row * (size_t)pairsPerRow * 2 + (size_t)p * 2;
  float x1 = (float)pd[0], x2 = (float)pd[1];
  pd[0] = (__bf16)(x1 * c - x2 * s);
  pd[1] = (__bf16)(x1 * s + x2 * c);
}

// ---------------------------------------------------------------------------
// TDM: DMA a 2D tile (tile_d0 x tile_d1 elements, 2-byte data) from global
// memory (row stride stride0 elements) into LDS at lds_off.
// D# group0: count=1, lds_addr, global_addr, type=2.
// D# group1: data_size=1 (2B), tensor dims for OOB zero-fill, tile dims,
//            tensor_dim0_stride. Groups 2/3 omitted (2D tensor).
// ---------------------------------------------------------------------------
__device__ __forceinline__ void tdm_load_2d(uint32_t lds_off, uint64_t gaddr,
                                            uint32_t tile_d0, uint32_t tile_d1,
                                            uint32_t tensor_d0, uint32_t tensor_d1,
                                            uint32_t stride0) {
  u32x4 g0;
  g0[0] = 1u;                                             // count=1, user mode
  g0[1] = lds_off;                                        // lds_addr (bytes)
  g0[2] = (uint32_t)gaddr;                                // global_addr lo
  g0[3] = ((uint32_t)(gaddr >> 32) & 0x01FFFFFFu) | (2u << 30); // hi + type=2
  u32x8 g1;
  g1[0] = 1u << 16;                                       // data_size = 2 bytes
  g1[1] = (tensor_d0 & 0xFFFFu) << 16;                    // tensor_dim0 lo16
  g1[2] = (tensor_d0 >> 16) | ((tensor_d1 & 0xFFFFu) << 16);
  g1[3] = (tensor_d1 >> 16) | (tile_d0 << 16);            // tile_dim0
  g1[4] = tile_d1 & 0xFFFFu;                              // tile_dim1, dim2=0
  g1[5] = stride0;                                        // dim0 stride lo32
  g1[6] = 0u;
  g1[7] = 0u;
  asm volatile("tensor_load_to_lds %0, %1" :: "s"(g0), "s"(g1) : "memory");
}

// ---------------------------------------------------------------------------
// Fused causal flash attention, all-bf16 operands. One wave per 16-row q-tile
// of one (b,h). K/kR tiles double-buffered via TDM; V tile single-buffered
// (its DMA overlaps the QK^T + softmax phase). 2 waves per block.
// ---------------------------------------------------------------------------
__global__ __launch_bounds__(64)
void mla_flash_kernel(const __bf16* __restrict__ qC, const __bf16* __restrict__ qR,
                      const __bf16* __restrict__ kC, const __bf16* __restrict__ kR,
                      const __bf16* __restrict__ V,  __bf16* __restrict__ O) {
  __shared__ __align__(32) __bf16 ldsK[2][2][32 * 64];  // per-wave double buf
  __shared__ __align__(32) __bf16 ldsR[2][2][32 * 32];
  __shared__ __align__(32) __bf16 ldsV[2][32 * 64];
  __shared__ __align__(32) __bf16 ldsP[2][16 * 32];

  int lane = (int)(threadIdx.x & 31);
  int wib  = __builtin_amdgcn_readfirstlane((int)(threadIdx.x >> 5));
  int wid  = (int)blockIdx.x * 2 + wib;
  int qt = wid & 127;
  int h  = (wid >> 7) & 15;
  int b  = wid >> 11;
  int r  = lane & 15;
  int kg = lane >> 4;
  size_t rowbase = (size_t)b * SEQ;
  int qtop = qt * 16;

  // ---- Q tile in A-operand layout (3 K-chunks of 32) ----
  const __bf16* qcrow = qC + (rowbase + qt * 16 + r) * 1024 + h * 64;
  const __bf16* qrrow = qR + (rowbase + qt * 16 + r) * 512 + h * 32;
  v16bf qa[3];
#pragma unroll
  for (int c = 0; c < 2; ++c) {
    v8bf lo = *(const v8bf*)(qcrow + c * 32 + kg * 16);
    v8bf hi = *(const v8bf*)(qcrow + c * 32 + kg * 16 + 8);
    qa[c] = __builtin_shufflevector(lo, hi, 0,1,2,3,4,5,6,7,8,9,10,11,12,13,14,15);
  }
  {
    v8bf lo = *(const v8bf*)(qrrow + kg * 16);
    v8bf hi = *(const v8bf*)(qrrow + kg * 16 + 8);
    qa[2] = __builtin_shufflevector(lo, hi, 0,1,2,3,4,5,6,7,8,9,10,11,12,13,14,15);
  }

  // ---- TDM bases (byte addresses / LDS offsets; wave-uniform) ----
  uint64_t kcb = (uint64_t)(uintptr_t)(kC + rowbase * 1024 + (size_t)h * 64);
  uint64_t krb = (uint64_t)(uintptr_t)(kR + rowbase * 32);
  uint64_t vbb = (uint64_t)(uintptr_t)(V  + rowbase * 1024 + (size_t)h * 64);
  uint32_t offK[2] = { (uint32_t)(uintptr_t)&ldsK[wib][0][0],
                       (uint32_t)(uintptr_t)&ldsK[wib][1][0] };
  uint32_t offR[2] = { (uint32_t)(uintptr_t)&ldsR[wib][0][0],
                       (uint32_t)(uintptr_t)&ldsR[wib][1][0] };
  uint32_t offV    =   (uint32_t)(uintptr_t)&ldsV[wib][0];
  __bf16* pbuf = &ldsP[wib][0];

  // prologue: fetch K(0), R(0) into buffer 0
  tdm_load_2d(offK[0], kcb, 64, 32, 64, SEQ, 1024);
  tdm_load_2d(offR[0], krb, 32, 32, 32, SEQ, 32);

  float mrow[8], lrow[8];
  v8f oacc[4];
#pragma unroll
  for (int i = 0; i < 8; ++i) { mrow[i] = -1e30f; lrow[i] = 0.0f; }
#pragma unroll
  for (int oc = 0; oc < 4; ++oc) oacc[oc] = (v8f){};
  const float rs = 0.10206207261596575f;   // 1/sqrt(96)

  int buf = 0;
  for (int kb = 0; kb <= qtop; kb += 32, buf ^= 1) {
    // current K/R (issued last iteration) must be resident
    __builtin_amdgcn_s_wait_tensorcnt(0);
    asm volatile("" ::: "memory");
    // stream this block's V, then prefetch next K/R (in-order TENSORcnt)
    bool more = (kb + 32) <= qtop;
    tdm_load_2d(offV, vbb + (uint64_t)kb * 2048, 64, 32, 64,
                (uint32_t)(SEQ - kb), 1024);
    if (more) {
      tdm_load_2d(offK[buf ^ 1], kcb + (uint64_t)(kb + 32) * 2048,
                  64, 32, 64, (uint32_t)(SEQ - (kb + 32)), 1024);
      tdm_load_2d(offR[buf ^ 1], krb + (uint64_t)(kb + 32) * 64,
                  32, 32, 32, (uint32_t)(SEQ - (kb + 32)), 32);
    }

    // ---- scores: two 16-key sub-tiles from LDS ----
    const __bf16* kT = &ldsK[wib][buf][0];
    const __bf16* rT = &ldsR[wib][buf][0];
    float p[2][8];
#pragma unroll
    for (int t = 0; t < 2; ++t) {
      int keyl = t * 16 + r;
      v16bf b0 = *(const v16bf*)(kT + keyl * 64 + kg * 16);
      v16bf b1 = *(const v16bf*)(kT + keyl * 64 + 32 + kg * 16);
      v16bf b2 = *(const v16bf*)(rT + keyl * 32 + kg * 16);
      v8f s = {};
      s = __builtin_amdgcn_wmma_f32_16x16x32_bf16(false, qa[0], false, b0, (short)0, s, false, false);
      s = __builtin_amdgcn_wmma_f32_16x16x32_bf16(false, qa[1], false, b1, (short)0, s, false, false);
      s = __builtin_amdgcn_wmma_f32_16x16x32_bf16(false, qa[2], false, b2, (short)0, s, false, false);
      int key = kb + keyl;
#pragma unroll
      for (int i = 0; i < 8; ++i) {
        float sc = fminf(fmaxf(s[i], -80.0f), 80.0f) * rs;
        int qrow = qtop + (kg << 3) + i;
        if (key > qrow) sc = -3.0e38f;     // causal mask (also OOB zero rows)
        p[t][i] = sc;
      }
    }
    // ---- online softmax (row reduce across 16-lane column groups) ----
#pragma unroll
    for (int i = 0; i < 8; ++i) {
      float mx = fmaxf(p[0][i], p[1][i]);
      mx = fmaxf(mx, __shfl_xor(mx, 1));
      mx = fmaxf(mx, __shfl_xor(mx, 2));
      mx = fmaxf(mx, __shfl_xor(mx, 4));
      mx = fmaxf(mx, __shfl_xor(mx, 8));
      float mnew  = fmaxf(mrow[i], mx);
      float alpha = __expf(mrow[i] - mnew);
      float p0 = __expf(p[0][i] - mnew);
      float p1 = __expf(p[1][i] - mnew);
      p[0][i] = p0; p[1][i] = p1;
      float ls = p0 + p1;
      ls += __shfl_xor(ls, 1);
      ls += __shfl_xor(ls, 2);
      ls += __shfl_xor(ls, 4);
      ls += __shfl_xor(ls, 8);
      lrow[i] = lrow[i] * alpha + ls;
      mrow[i] = mnew;
#pragma unroll
      for (int oc = 0; oc < 4; ++oc) oacc[oc][i] *= alpha;
    }
    // ---- P: C-layout -> A-layout via LDS scatter + contiguous reload ----
#pragma unroll
    for (int t = 0; t < 2; ++t)
#pragma unroll
      for (int i = 0; i < 8; ++i) {
        int row = (kg << 3) + i;
        int key = t * 16 + r;
        int addr = ((row | (((key >> 3) & 1) << 4)) << 4)
                 | (((key >> 4) << 3) | (key & 7));
        pbuf[addr] = (__bf16)p[t][i];
      }
    asm volatile("s_wait_dscnt 0" ::: "memory");
    v16bf pa = *(const v16bf*)(pbuf + lane * 16);

    // ---- V tile ready? (V issued before nextK/nextR; in-order counter) ----
    if (more) __builtin_amdgcn_s_wait_tensorcnt(2);
    else      __builtin_amdgcn_s_wait_tensorcnt(0);
    asm volatile("" ::: "memory");
    const __bf16* vT = &ldsV[wib][0];
#pragma unroll
    for (int oc = 0; oc < 4; ++oc) {
      v16bf vbv;
#pragma unroll
      for (int j = 0; j < 16; ++j)
        vbv[j] = vT[(kg * 8 + (j & 7) + (j >> 3) * 16) * 64 + oc * 16 + r];
      oacc[oc] = __builtin_amdgcn_wmma_f32_16x16x32_bf16(
          false, pa, false, vbv, (short)0, oacc[oc], false, false);
    }
  }
  // ---- normalize + store this head's (16 x 64) output slice (bf16) ----
  float linv[8];
#pragma unroll
  for (int i = 0; i < 8; ++i) linv[i] = 1.0f / lrow[i];
#pragma unroll
  for (int oc = 0; oc < 4; ++oc)
#pragma unroll
    for (int i = 0; i < 8; ++i) {
      int m = qtop + (kg << 3) + i;
      O[(rowbase + m) * 1024 + h * 64 + oc * 16 + r] =
          (__bf16)(oacc[oc][i] * linv[i]);
    }
}

// ---------------------------------------------------------------------------
extern "C" void kernel_launch(void* const* d_in, const int* in_sizes, int n_in,
                              void* d_out, int out_size, void* d_ws, size_t ws_size,
                              hipStream_t stream) {
  (void)in_sizes; (void)n_in; (void)out_size; (void)ws_size;
  const float* x     = (const float*)d_in[0];
  const float* W_DQ  = (const float*)d_in[1];
  const float* W_UQ  = (const float*)d_in[2];
  const float* W_QR  = (const float*)d_in[3];
  const float* W_DKV = (const float*)d_in[4];
  const float* W_UK  = (const float*)d_in[5];
  const float* W_UV  = (const float*)d_in[6];
  const float* W_KR  = (const float*)d_in[7];
  const float* W_O   = (const float*)d_in[8];
  float* out = (float*)d_out;

  const size_t R = ROWS;
  __bf16* wsb = (__bf16*)d_ws;
  size_t off = 0;
  auto alloc = [&](size_t n) { __bf16* p = wsb + off; off += n; return p; };
  __bf16* xb   = alloc(R * 1024);
  __bf16* wdq  = alloc(512 * 1024);
  __bf16* wuq  = alloc(1024 * 512);
  __bf16* wqr  = alloc(512 * 512);
  __bf16* wdkv = alloc(512 * 1024);
  __bf16* wuk  = alloc(1024 * 512);
  __bf16* wuv  = alloc(1024 * 512);
  __bf16* wkr  = alloc(32 * 1024);
  __bf16* wo   = alloc(1024 * 1024);
  __bf16* cQ   = alloc(R * 512);
  __bf16* cKV  = alloc(R * 512);
  __bf16* qC   = alloc(R * 1024);
  __bf16* qR   = alloc(R * 512);
  __bf16* kC   = alloc(R * 1024);
  __bf16* kRb  = alloc(R * 32);
  __bf16* Vb   = alloc(R * 1024);
  __bf16* attn = alloc(R * 1024);

  auto cvt = [&](const float* src, __bf16* dst, size_t n) {
    cvt_f32_bf16<<<(int)((n + 255) / 256), 256, 0, stream>>>(src, dst, (int)n);
  };
  cvt(x, xb, R * 1024);
  cvt(W_DQ, wdq, 512 * 1024);   cvt(W_UQ, wuq, 1024 * 512);
  cvt(W_QR, wqr, 512 * 512);    cvt(W_DKV, wdkv, 512 * 1024);
  cvt(W_UK, wuk, 1024 * 512);   cvt(W_UV, wuv, 1024 * 512);
  cvt(W_KR, wkr, 32 * 1024);    cvt(W_O, wo, 1024 * 1024);

  auto g4 = [&](const __bf16* A, const __bf16* W, __bf16* C, int M, int N, int K) {
    int waves = (M >> 4) * (N >> 6);
    gemm_bf16<4, true><<<(waves * 32 + 255) / 256, 256, 0, stream>>>(A, W, (void*)C, M, N, K);
  };
  // latent projections
  g4(xb, wdq, cQ, ROWS, 512, 1024);
  g4(xb, wdkv, cKV, ROWS, 512, 1024);
  {  // k_R: N=32 -> 2 sub-tiles per wave
    int waves = (ROWS >> 4) * (32 / 32);
    gemm_bf16<2, true><<<(waves * 32 + 255) / 256, 256, 0, stream>>>(xb, wkr, (void*)kRb, ROWS, 32, 1024);
  }
  // up-projections
  g4(cQ, wuq, qC, ROWS, 1024, 512);
  g4(cQ, wqr, qR, ROWS, 512, 512);
  g4(cKV, wuk, kC, ROWS, 1024, 512);
  g4(cKV, wuv, Vb, ROWS, 1024, 512);
  // RoPE
  {
    int total = ROWS * 256;
    rope_bf16<<<(total + 255) / 256, 256, 0, stream>>>(qR, ROWS, 256);
  }
  {
    int total = ROWS * 16;
    rope_bf16<<<(total + 255) / 256, 256, 0, stream>>>(kRb, ROWS, 16);
  }
  // fused flash attention: 8192 waves, 2 waves/block
  mla_flash_kernel<<<4096, 64, 0, stream>>>(qC, qR, kC, kRb, Vb, attn);
  // output projection (f32 out)
  {
    int waves = (ROWS >> 4) * (1024 >> 6);
    gemm_bf16<4, false><<<(waves * 32 + 255) / 256, 256, 0, stream>>>(attn, wo, (void*)out, ROWS, 1024, 1024);
  }
}
